// Multi_Head_Att_35596688949300
// MI455X (gfx1250) — matrix operand order repaired and checked by reading, not compiled, not run on hardware
//
#include <hip/hip_runtime.h>
#include <hip/hip_bf16.h>

typedef __attribute__((ext_vector_type(16))) _Float16 v16h;
typedef __attribute__((ext_vector_type(8)))  _Float16 v8h;
typedef __attribute__((ext_vector_type(8)))  float    v8f;
typedef __attribute__((ext_vector_type(4)))  int      v4i;

#define HIDDEN 1024
#define TOKENS 16384   // 4 * 4096
#define CHUNK  256     // 4096 / 16 heads
#define KS     40      // LDS stride (halves) for 32-wide k tiles: 80B = 16B multiple
#define PS     264     // LDS stride (halves) for 256-wide key rows: 528B = 16B multiple

// ---------------- CDNA5 async global->LDS (ASYNCcnt) --------------------------
#if defined(__gfx1250__) && __has_builtin(__builtin_amdgcn_global_load_async_to_lds_b128)
#define HAS_ASYNC_LDS 1
#endif

typedef __attribute__((address_space(1))) v4i gv4i;   // global 16B vector
typedef __attribute__((address_space(3))) v4i lv4i;   // LDS    16B vector

__device__ __forceinline__ void async_copy16(const _Float16* g, _Float16* l) {
#ifdef HAS_ASYNC_LDS
  __builtin_amdgcn_global_load_async_to_lds_b128((gv4i*)g, (lv4i*)l, 0, 0);
#else
  *(float4*)l = *(const float4*)g;
#endif
}

template <int N>
__device__ __forceinline__ void wait_async_le() {
#ifdef HAS_ASYNC_LDS
#if __has_builtin(__builtin_amdgcn_s_wait_asynccnt)
  __builtin_amdgcn_s_wait_asynccnt(N);
#else
  asm volatile("s_wait_asynccnt %0" ::"i"(N) : "memory");
#endif
#endif
}

// ---------------- WMMA fragment builders (per ISA VGPR layouts) ---------------
union FragU { v16h v; unsigned u[8]; };

// A 16x32 f16: lane half h -> K {8h..8h+7} in v0-3, {16+8h..23+8h} in v4-7; (k,k+1) per dword.
__device__ __forceinline__ v16h load_frag_a(const _Float16* row, int half) {
  FragU f;
  int klo = half * 8;
#pragma unroll
  for (int j = 0; j < 4; ++j) {
    f.u[j]     = *(const unsigned*)(row + klo + 2 * j);
    f.u[4 + j] = *(const unsigned*)(row + 16 + klo + 2 * j);
  }
  return f.v;
}

// B 32x16 f16: lane half h -> K {8h..8h+7} in v0-3, {16+8h..23+8h} in v4-7; tile stored n-major so (k,k+1) adjacent.
__device__ __forceinline__ v16h load_frag_b(const _Float16* row, int half) {
  FragU f;
  int k0 = half * 8;
#pragma unroll
  for (int j = 0; j < 4; ++j) {
    f.u[j]     = *(const unsigned*)(row + k0 + 2 * j);
    f.u[4 + j] = *(const unsigned*)(row + 16 + k0 + 2 * j);
  }
  return f.v;
}

__device__ __forceinline__ v8f wmma_f16(v16h a, v16h b, v8f c) {
  return __builtin_amdgcn_wmma_f32_16x16x32_f16(false, a, false, b, (short)0, c, false, false);
}

// ---------------------------------------------------------------------------
// Kernel 0: fp32 -> f16 bulk convert (8 elems/thread, 2x b128 load, 1x b128 store)
// ---------------------------------------------------------------------------
__global__ __launch_bounds__(256) void cvt_kernel(const float* __restrict__ src,
                                                  _Float16* __restrict__ dst, int n8) {
  int i = blockIdx.x * 256 + threadIdx.x;
  if (i >= n8) return;
  const float4* s = (const float4*)src + (size_t)i * 2;
  float4 f0 = s[0], f1 = s[1];
  v8h h;
  h[0] = (_Float16)f0.x; h[1] = (_Float16)f0.y; h[2] = (_Float16)f0.z; h[3] = (_Float16)f0.w;
  h[4] = (_Float16)f1.x; h[5] = (_Float16)f1.y; h[6] = (_Float16)f1.z; h[7] = (_Float16)f1.w;
  *(v8h*)(dst + (size_t)i * 8) = h;
}

// ---------------------------------------------------------------------------
// Kernel 1: Q/K/V projection from pre-converted f16 x and W.
// Block tile 64x128, 4 waves (2x2), wave tile 32x64: 8 wmma per 6 frag loads.
// Double-buffered async global->LDS staging, fixed-trip unrolled staging but
// NON-unrolled pipeline loop (keeps accumulators in place, no hazard NOPs).
// Q,K stored [token][e]; V stored TRANSPOSED [e][token].
// ---------------------------------------------------------------------------
__global__ __launch_bounds__(128) void qkv_kernel(
    const _Float16* __restrict__ xh, const _Float16* __restrict__ wh,
    const float* __restrict__ bq, const float* __restrict__ bk,
    const float* __restrict__ bv, _Float16* __restrict__ qkv) {
  __shared__ _Float16 lds_a[2][64 * KS];    // [row m][k]  (x tile)
  __shared__ _Float16 lds_b[2][128 * KS];   // [col e][k]  (W tile)

  const int t = threadIdx.x;
  const int lane = t & 31, half = lane >> 4, ln = lane & 15;
  const int wave = t >> 5, wm = wave >> 1, wn = wave & 1;

  const int m0  = blockIdx.x * 64;
  const int mat = blockIdx.y >> 3;          // 0=Q 1=K 2=V
  const int n0  = (blockIdx.y & 7) * 128;

  const _Float16* W    = wh + (size_t)mat * HIDDEN * HIDDEN;
  const float*    bias = (mat == 0) ? bq : (mat == 1) ? bk : bv;

  const int rA = t >> 2, c8A = (t & 3) * 8;   // staging coords: i = t + ii*128

  auto stage = [&](int buf, int k0) {
#pragma unroll
    for (int ii = 0; ii < 2; ++ii) {          // x: 64 rows x 4 16B chunks = 2/thread
      int r = rA + ii * 32;
      async_copy16(&xh[(size_t)(m0 + r) * HIDDEN + k0 + c8A], &lds_a[buf][r * KS + c8A]);
    }
#pragma unroll
    for (int ii = 0; ii < 4; ++ii) {          // W: 128 rows x 4 chunks = 4/thread
      int r = rA + ii * 32;
      async_copy16(&W[(size_t)(n0 + r) * HIDDEN + k0 + c8A], &lds_b[buf][r * KS + c8A]);
    }
  };

  v8f acc[2][4] = {};
  stage(0, 0);
#pragma unroll 1
  for (int kt = 0; kt < 32; ++kt) {
    int cur = kt & 1;
    if (kt + 1 < 32) { stage(cur ^ 1, (kt + 1) * 32); wait_async_le<6>(); }
    else             { wait_async_le<0>(); }
    __syncthreads();
    v16h a0 = load_frag_a(&lds_a[cur][(wm * 32 + 0 + ln) * KS], half);
    v16h a1 = load_frag_a(&lds_a[cur][(wm * 32 + 16 + ln) * KS], half);
#pragma unroll
    for (int j = 0; j < 4; ++j) {
      v16h b = load_frag_b(&lds_b[cur][(wn * 64 + j * 16 + ln) * KS], half);
      acc[0][j] = wmma_f16(a0, b, acc[0][j]);
      acc[1][j] = wmma_f16(a1, b, acc[1][j]);
    }
    __syncthreads();
  }

  _Float16* dst = qkv + (size_t)mat * TOKENS * HIDDEN;
#pragma unroll
  for (int i = 0; i < 2; ++i)
#pragma unroll
    for (int j = 0; j < 4; ++j) {
      int col = n0 + wn * 64 + j * 16 + ln;
      float bb = bias[col];
#pragma unroll
      for (int r = 0; r < 8; ++r) {            // C layout: m = 8*half + r, n = ln
        int rowg = m0 + wm * 32 + i * 16 + 8 * half + r;
        _Float16 val = (_Float16)(acc[i][j][r] + bb);
        if (mat == 2)
          dst[(size_t)col * TOKENS + rowg] = val;        // V transposed: [e][token]
        else
          dst[(size_t)rowg * HIDDEN + col] = val;        // Q,K: [token][e]
      }
    }
}

// ---------------------------------------------------------------------------
// Kernel 2: per-chunk attention. Block = (chunk, 64-row query slab), 4 waves.
// Wave owns a 16x256 score strip -> softmax in-wave via shfl_xor over 16-lane
// halves. Double-buffered async staging for Q/K tiles and V column tiles.
// ---------------------------------------------------------------------------
__global__ __launch_bounds__(128) void attn_kernel(const _Float16* __restrict__ qkv,
                                                   float* __restrict__ out) {
  __shared__ _Float16 lds_q[2][64 * KS];    // [q row][k]
  __shared__ _Float16 lds_k[2][256 * KS];   // [key row][k]   (B of Q@K^T)
  __shared__ _Float16 lds_p[64 * PS];       // [q row][key]   (A of P@V)
  __shared__ _Float16 lds_v[2][16 * PS];    // [e col][key]   (B of P@V)

  const int t = threadIdx.x;
  const int lane = t & 31, half = lane >> 4, ln = lane & 15;
  const int wave = t >> 5;

  const int c0 = blockIdx.y * CHUNK;          // chunk token base
  const int m0 = c0 + blockIdx.x * 64;        // query slab base

  const _Float16* Q  = qkv;                                   // [token][e]
  const _Float16* K  = qkv + (size_t)TOKENS * HIDDEN;         // [token][e]
  const _Float16* Vt = qkv + 2 * (size_t)TOKENS * HIDDEN;     // [e][token]

  const int rS = t >> 2, c8S = (t & 3) * 8;   // k-tile staging coords
  const int nV = t >> 5, c8V = (t & 31) * 8;  // V-tile staging coords

  auto stageS = [&](int buf, int k0) {        // 2 + 8 = 10 async ops per thread
#pragma unroll
    for (int ii = 0; ii < 2; ++ii) {
      int r = rS + ii * 32;
      async_copy16(&Q[(size_t)(m0 + r) * HIDDEN + k0 + c8S], &lds_q[buf][r * KS + c8S]);
    }
#pragma unroll
    for (int ii = 0; ii < 8; ++ii) {
      int r = rS + ii * 32;
      async_copy16(&K[(size_t)(c0 + r) * HIDDEN + k0 + c8S], &lds_k[buf][r * KS + c8S]);
    }
  };
  auto stageV = [&](int buf, int nt) {        // 4 async ops per thread
#pragma unroll
    for (int ii = 0; ii < 4; ++ii) {
      int n = nV + ii * 4;
      async_copy16(&Vt[(size_t)(nt * 16 + n) * TOKENS + c0 + c8V], &lds_v[buf][n * PS + c8V]);
    }
  };

  // ---- S = Q @ K^T ----
  v8f sacc[16] = {};
  stageS(0, 0);
#pragma unroll 1
  for (int kt = 0; kt < 32; ++kt) {
    int cur = kt & 1;
    if (kt + 1 < 32) { stageS(cur ^ 1, (kt + 1) * 32); wait_async_le<10>(); }
    else             { wait_async_le<0>(); }
    __syncthreads();
    v16h a = load_frag_a(&lds_q[cur][(wave * 16 + ln) * KS], half);
#pragma unroll
    for (int j = 0; j < 16; ++j) {
      v16h b = load_frag_b(&lds_k[cur][(j * 16 + ln) * KS], half);
      sacc[j] = wmma_f16(a, b, sacc[j]);
    }
    __syncthreads();
  }

  // ---- softmax over 256 keys; row m = wave*16 + 8*half + r ----
#pragma unroll
  for (int r = 0; r < 8; ++r) {
    float mx = sacc[0][r];
#pragma unroll
    for (int j = 1; j < 16; ++j) mx = fmaxf(mx, sacc[j][r]);
#pragma unroll
    for (int m = 1; m < 16; m <<= 1) mx = fmaxf(mx, __shfl_xor(mx, m, 32));
    float s = 0.f;
#pragma unroll
    for (int j = 0; j < 16; ++j) { float e = __expf(sacc[j][r] - mx); sacc[j][r] = e; s += e; }
#pragma unroll
    for (int m = 1; m < 16; m <<= 1) s += __shfl_xor(s, m, 32);
    float inv = 1.f / s;
    int prow = wave * 16 + 8 * half + r;
#pragma unroll
    for (int j = 0; j < 16; ++j)
      lds_p[prow * PS + j * 16 + ln] = (_Float16)(sacc[j][r] * inv);
  }
  __syncthreads();

  // ---- O = P @ V ; hoist all 8 A-fragments of this wave's 16 rows ----
  stageV(0, 0);
  v16h pa[8];
#pragma unroll
  for (int kk = 0; kk < 8; ++kk)
    pa[kk] = load_frag_a(lds_p + (wave * 16 + ln) * PS + 32 * kk, half);

#pragma unroll 1
  for (int nt = 0; nt < 64; ++nt) {
    int cur = nt & 1;
    if (nt + 1 < 64) { stageV(cur ^ 1, nt + 1); wait_async_le<4>(); }
    else             { wait_async_le<0>(); }
    __syncthreads();
    v8f oacc = {};
#pragma unroll
    for (int kk = 0; kk < 8; ++kk) {
      v16h b = load_frag_b(&lds_v[cur][ln * PS + 32 * kk], half);
      oacc = wmma_f16(pa[kk], b, oacc);
    }
#pragma unroll
    for (int r = 0; r < 8; ++r)
      out[(size_t)(m0 + wave * 16 + 8 * half + r) * HIDDEN + nt * 16 + ln] = oacc[r];
    __syncthreads();
  }
}

extern "C" void kernel_launch(void* const* d_in, const int* in_sizes, int n_in,
                              void* d_out, int out_size, void* d_ws, size_t ws_size,
                              hipStream_t stream) {
  const float* x  = (const float*)d_in[0];
  const float* Wq = (const float*)d_in[1];
  const float* bq = (const float*)d_in[2];
  const float* Wk = (const float*)d_in[3];
  const float* bk = (const float*)d_in[4];
  const float* Wv = (const float*)d_in[5];
  const float* bv = (const float*)d_in[6];

  // ws layout (halves): [QKV: 3*T*H][xh: T*H][wh: 3*H*H]  ~= 140.5 MB
  _Float16* qkv = (_Float16*)d_ws;
  _Float16* xh  = qkv + 3 * (size_t)TOKENS * HIDDEN;
  _Float16* wh  = xh + (size_t)TOKENS * HIDDEN;
  float* out = (float*)d_out;

  const int xn8 = TOKENS * HIDDEN / 8, wn8 = HIDDEN * HIDDEN / 8;
  cvt_kernel<<<(xn8 + 255) / 256, 256, 0, stream>>>(x, xh, xn8);
  cvt_kernel<<<(wn8 + 255) / 256, 256, 0, stream>>>(Wq, wh, wn8);
  cvt_kernel<<<(wn8 + 255) / 256, 256, 0, stream>>>(Wk, wh + (size_t)HIDDEN * HIDDEN, wn8);
  cvt_kernel<<<(wn8 + 255) / 256, 256, 0, stream>>>(Wv, wh + 2 * (size_t)HIDDEN * HIDDEN, wn8);

  qkv_kernel<<<dim3(TOKENS / 64, 24), 128, 0, stream>>>(xh, wh, bq, bk, bv, qkv);
  attn_kernel<<<dim3(4, 64), 128, 0, stream>>>(qkv, out);
}
